// Transformer_63273458205428
// MI455X (gfx1250) — matrix operand order repaired and checked by reading, not compiled, not run on hardware
//
#include <hip/hip_runtime.h>
#include <hip/hip_bf16.h>
#include <math.h>

// ---------------------------------------------------------------------------
// GPT-2-style forward pass for gfx1250 (MI455X).
// S=2048, D=768, H=12, hs=64, L=4, V=32000.
// GEMMs: v_wmma_f32_16x16x32_f16, fp16 operands, fp32 accumulate.
// Double-buffered LDS tiles staged with GLOBAL_LOAD_ASYNC_TO_LDS_B128 so the
// async engine copies tile k+1 underneath tile k's WMMAs.
// ---------------------------------------------------------------------------

typedef __attribute__((ext_vector_type(16))) _Float16 v16h;
typedef __attribute__((ext_vector_type(8)))  _Float16 v8h;
typedef __attribute__((ext_vector_type(8)))  float    v8f;
typedef __attribute__((ext_vector_type(4)))  int      v4i_;

#if __has_builtin(__builtin_amdgcn_global_load_async_to_lds_b128)
#define HAVE_ASYNC_LDS 1
#if __has_builtin(__builtin_amdgcn_s_wait_asynccnt)
#define ASYNC_WAIT() __builtin_amdgcn_s_wait_asynccnt(0)
#else
#define ASYNC_WAIT() asm volatile("s_wait_asynccnt 0x0" ::: "memory")
#endif
__device__ __forceinline__ void async_copy16(const _Float16* gp, _Float16* lp) {
    // builtin signature: arg0 = int4 addrspace(1)*, arg1 = int4 addrspace(3)*,
    // then imm offset, imm cpol.
    __builtin_amdgcn_global_load_async_to_lds_b128(
        (__attribute__((address_space(1))) v4i_*)(void*)const_cast<_Float16*>(gp),
        (__attribute__((address_space(3))) v4i_*)(void*)lp,
        0, 0);
}
#else
#define HAVE_ASYNC_LDS 0
#define ASYNC_WAIT()
__device__ __forceinline__ void async_copy16(const _Float16* gp, _Float16* lp) {
    *(v8h*)lp = *(const v8h*)gp;
}
#endif

// ------------------------------ reductions ---------------------------------

__device__ __forceinline__ float waveReduceSum(float v) {
    #pragma unroll
    for (int off = 16; off > 0; off >>= 1) v += __shfl_xor(v, off, 32);
    return v;
}
__device__ __forceinline__ float waveReduceMax(float v) {
    #pragma unroll
    for (int off = 16; off > 0; off >>= 1) v = fmaxf(v, __shfl_xor(v, off, 32));
    return v;
}
// blockDim.x == 256 (8 waves) assumed.
__device__ __forceinline__ float blockReduceSum(float v, float* sred) {
    int lane = threadIdx.x & 31, w = threadIdx.x >> 5;
    v = waveReduceSum(v);
    __syncthreads();
    if (lane == 0) sred[w] = v;
    __syncthreads();
    float tot = 0.f;
    #pragma unroll
    for (int i = 0; i < 8; ++i) tot += sred[i];
    return tot;
}
__device__ __forceinline__ float blockReduceMax(float v, float* sred) {
    int lane = threadIdx.x & 31, w = threadIdx.x >> 5;
    v = waveReduceMax(v);
    __syncthreads();
    if (lane == 0) sred[w] = v;
    __syncthreads();
    float tot = -3.0e38f;
    #pragma unroll
    for (int i = 0; i < 8; ++i) tot = fmaxf(tot, sred[i]);
    return tot;
}

// ------------------------------ WMMA GEMM ----------------------------------
// C[M,N] = A[M,K] @ B (+ epilogue). A,B fp16. B is [K,N] row-major, or [N,K]
// if TRANSB. REQUIRES: M multiple of 128, N multiple of 64, K multiple of 64,
// leading dims / base pointers 16B-aligned.
// Block: 128 threads = 4 waves; block tile 128x64x64; wave tile 64x32
// (acc[4][2]); 16 WMMAs per wave per K-step; double-buffered async staging.

enum { EP_NONE = 0, EP_BIAS = 1, EP_BIAS_GELU = 2, EP_SCALE_MASK = 3 };

template <int EPI, bool TRANSB, bool OUTH>
__global__ __launch_bounds__(128)
void wmma_gemm_kernel(const _Float16* __restrict__ A, const _Float16* __restrict__ B,
                      const float* __restrict__ bias, void* __restrict__ Cv,
                      int M, int N, int K, int lda, int ldb, int ldc,
                      long aBatch, long bBatch, long cBatch, float scale)
{
    (void)M; (void)N;
    constexpr int BM = 128, BN = 64, BK = 64;
    constexpr int ASTR = BK + 8;   // 72 halfwords = 144B rows (16B-aligned)
    constexpr int BSTR = BK + 8;   // Bs stored [n][k]
    __shared__ __align__(16) _Float16 As[2 * BM * ASTR];
    __shared__ __align__(16) _Float16 Bs[2 * BN * BSTR];

    const long bz = blockIdx.z;
    A += bz * aBatch;  B += bz * bBatch;

    const int tileN = blockIdx.x * BN;
    const int tileM = blockIdx.y * BM;
    const int tid   = threadIdx.x;
    const int lane  = tid & 31;
    const int wave  = tid >> 5;
    const int half  = lane >> 4;
    const int mr    = lane & 15;
    const int waveM = (wave >> 1) * 64;
    const int waveN = (wave & 1)  * 32;

    v8f acc[4][2] = {};

    // Stage K-tile kt into LDS buffer `buf` (async where possible).
    auto stage = [&](int kt, int buf) {
        const int k0 = kt * BK;
        _Float16* Asb = As + buf * (BM * ASTR);
        _Float16* Bsb = Bs + buf * (BN * BSTR);
        // A tile: 128 x 64 halves = 1024 16B vectors, 8 per thread
        #pragma unroll
        for (int p = 0; p < 8; ++p) {
            int vec = tid + p * 128;
            int r = vec >> 3, c8 = (vec & 7) << 3;
            async_copy16(&A[(long)(tileM + r) * lda + k0 + c8], &Asb[r * ASTR + c8]);
        }
        if constexpr (TRANSB) {
            // B stored [N][K]: contiguous k, async copy
            #pragma unroll
            for (int p = 0; p < 4; ++p) {
                int vec = tid + p * 128;
                int r = vec >> 3, c8 = (vec & 7) << 3;
                async_copy16(&B[(long)(tileN + r) * ldb + k0 + c8], &Bsb[r * BSTR + c8]);
            }
        } else {
            // B stored [K][N]: coalesced global read along n, scatter to [n][k]
            #pragma unroll
            for (int p = 0; p < 4; ++p) {
                int vec = tid + p * 128;
                int kr = vec >> 3, c8 = (vec & 7) << 3;
                v8h d = *(const v8h*)&B[(long)(k0 + kr) * ldb + tileN + c8];
                #pragma unroll
                for (int e = 0; e < 8; ++e) Bsb[(c8 + e) * BSTR + kr] = d[e];
            }
        }
    };

    // MACs for the tile resident in buffer `buf`.
    auto compute = [&](int buf) {
        const _Float16* Asb = As + buf * (BM * ASTR);
        const _Float16* Bsb = Bs + buf * (BN * BSTR);
        #pragma unroll
        for (int c = 0; c < 2; ++c) {          // two K=32 sub-chunks
            v16h bf[2];                        // reused across the 4 A subtiles
            #pragma unroll
            for (int j = 0; j < 2; ++j) {
                const int brow = (waveN + j * 16 + mr) * BSTR + c * 32 + (half << 3);
                v8h b0 = *(const v8h*)&Bsb[brow];
                v8h b1 = *(const v8h*)&Bsb[brow + 16];
                #pragma unroll
                for (int e = 0; e < 8; ++e) { bf[j][e] = b0[e]; bf[j][e + 8] = b1[e]; }
            }
            #pragma unroll
            for (int i = 0; i < 4; ++i) {
                const int arow = (waveM + i * 16 + mr) * ASTR + c * 32;
                v8h a0 = *(const v8h*)&Asb[arow + (half << 3)];
                v8h a1 = *(const v8h*)&Asb[arow + 16 + (half << 3)];
                v16h af;
                #pragma unroll
                for (int e = 0; e < 8; ++e) { af[e] = a0[e]; af[e + 8] = a1[e]; }
                #pragma unroll
                for (int j = 0; j < 2; ++j) {
                    acc[i][j] = __builtin_amdgcn_wmma_f32_16x16x32_f16(
                        false, af, false, bf[j], (short)0, acc[i][j], false, false);
                }
            }
        }
    };

    // Software pipeline: copies for tile k+1 run under tile k's WMMAs.
    const int NK = K / BK;
    stage(0, 0);
    for (int kt = 0; kt < NK; ++kt) {
        ASYNC_WAIT();          // tile kt resident (overlapped with compute kt-1)
        __syncthreads();       // all waves done reading buf[(kt+1)&1]
        if (kt + 1 < NK) stage(kt + 1, (kt + 1) & 1);
        compute(kt & 1);
    }

    // ---- store: D VGPR r holds (M = r + 8*half, N = mr) ----
    float*    Cf = (float*)Cv;
    _Float16* Ch = (_Float16*)Cv;
    #pragma unroll
    for (int i = 0; i < 4; ++i) {
        #pragma unroll
        for (int j = 0; j < 2; ++j) {
            #pragma unroll
            for (int r = 0; r < 8; ++r) {
                int gm = tileM + waveM + i * 16 + (half << 3) + r;
                int gn = tileN + waveN + j * 16 + mr;
                float v = acc[i][j][r];
                if constexpr (EPI == EP_BIAS) {
                    v += bias[gn];
                } else if constexpr (EPI == EP_BIAS_GELU) {
                    v += bias[gn];
                    v = 0.5f * v * (1.f + erff(v * 0.70710678118654752f));
                } else if constexpr (EPI == EP_SCALE_MASK) {
                    v = (gn <= gm) ? v * scale : -1e30f;
                }
                long off = bz * cBatch + (long)gm * ldc + gn;
                if constexpr (OUTH) Ch[off] = (_Float16)v;
                else                Cf[off] = v;
            }
        }
    }
}

// --------------------- fp32 -> fp16 transpose (weights) --------------------
// in: [K][N] fp32 (per grid.z batch), out: [N][K] fp16. K,N multiples of 32.
__global__ __launch_bounds__(256)
void convert_transpose_kernel(const float* __restrict__ in, _Float16* __restrict__ out,
                              int K, int N, long inStride, long outStride)
{
    __shared__ float tile[32][33];
    in  += (long)blockIdx.z * inStride;
    out += (long)blockIdx.z * outStride;
    int n0 = blockIdx.x * 32, k0 = blockIdx.y * 32;
    int tx = threadIdx.x & 31, ty = threadIdx.x >> 5;    // 32 x 8
    #pragma unroll
    for (int i = ty; i < 32; i += 8)
        tile[i][tx] = in[(long)(k0 + i) * N + n0 + tx];
    __syncthreads();
    #pragma unroll
    for (int i = ty; i < 32; i += 8)
        out[(long)(n0 + i) * K + k0 + tx] = (_Float16)tile[tx][i];
}

// --------------------------- row-wise kernels ------------------------------

__global__ __launch_bounds__(256)
void embed_posn_kernel(const int* __restrict__ x, const float* __restrict__ emb,
                       const float* __restrict__ posn, float* __restrict__ h,
                       _Float16* __restrict__ h16, int D)
{
    long s = blockIdx.x;
    int tok = x[s];
    for (int d = threadIdx.x; d < D; d += 256) {
        float v = emb[(long)tok * D + d] + posn[s * D + d];
        h[s * D + d]   = v;
        h16[s * D + d] = (_Float16)v;
    }
}

// in-place softmax on fp32 rows; optionally mirrors result to fp16.
__global__ __launch_bounds__(256)
void softmax_rows_kernel(float* __restrict__ data, _Float16* __restrict__ out16, int cols)
{
    __shared__ float sred[8];
    long r = blockIdx.x;
    float* p = data + r * (long)cols;
    float m = -3.0e38f;
    for (int c = threadIdx.x; c < cols; c += 256) m = fmaxf(m, p[c]);
    m = blockReduceMax(m, sred);
    float sum = 0.f;
    for (int c = threadIdx.x; c < cols; c += 256) {
        float e = expf(p[c] - m);
        p[c] = e;
        sum += e;
    }
    sum = blockReduceSum(sum, sred);
    float inv = 1.f / sum;
    for (int c = threadIdx.x; c < cols; c += 256) {
        float v = p[c] * inv;
        p[c] = v;
        if (out16) out16[r * (long)cols + c] = (_Float16)v;
    }
}

// h[row] += LayerNorm(x[row]) * g + b ; also writes fp16 mirror of h.
__global__ __launch_bounds__(256)
void ln_residual_kernel(float* __restrict__ h, _Float16* __restrict__ h16,
                        const float* __restrict__ x,
                        const float* __restrict__ g, const float* __restrict__ b, int D)
{
    __shared__ float sred[8];
    long s = blockIdx.x;
    const float* xr = x + s * (long)D;
    float*       hr = h + s * (long)D;
    _Float16*    hh = h16 + s * (long)D;

    float sum = 0.f;
    for (int d = threadIdx.x; d < D; d += 256) sum += xr[d];
    float mean = blockReduceSum(sum, sred) / (float)D;

    float vs = 0.f;
    for (int d = threadIdx.x; d < D; d += 256) { float t = xr[d] - mean; vs += t * t; }
    float inv = rsqrtf(blockReduceSum(vs, sred) / (float)D + 1e-5f);

    for (int d = threadIdx.x; d < D; d += 256) {
        float v = hr[d] + (xr[d] - mean) * inv * g[d] + b[d];
        hr[d] = v;
        hh[d] = (_Float16)v;
    }
}

// ------------------------------ launcher -----------------------------------

extern "C" void kernel_launch(void* const* d_in, const int* in_sizes, int n_in,
                              void* d_out, int out_size, void* d_ws, size_t ws_size,
                              hipStream_t stream)
{
    (void)in_sizes; (void)n_in; (void)out_size; (void)ws_size;

    constexpr int S = 2048, D = 768, H = 12, HS = 64, L = 4, V = 32000;
    constexpr int QN = 3 * HS;             // 192
    constexpr int D4 = 4 * D;              // 3072
    const float scale = 0.125f;            // 1/sqrt(64)

    const int*   x      = (const int*)  d_in[0];
    const float* embw   = (const float*)d_in[1];
    const float* posn   = (const float*)d_in[2];
    const float* ln1_g  = (const float*)d_in[3];
    const float* ln1_b  = (const float*)d_in[4];
    const float* qkv_w  = (const float*)d_in[5];   // [L,64,192]
    const float* out_w  = (const float*)d_in[6];   // [L,768,768]
    const float* ln2_g  = (const float*)d_in[7];
    const float* ln2_b  = (const float*)d_in[8];
    const float* w1     = (const float*)d_in[9];   // [L,768,3072]
    const float* b1     = (const float*)d_in[10];  // [L,3072]
    const float* w2     = (const float*)d_in[11];  // [L,3072,768]
    const float* b2     = (const float*)d_in[12];  // [L,768]
    const float* uw     = (const float*)d_in[13];  // [768,32000]
    const float* ub     = (const float*)d_in[14];  // [32000]
    float*       out    = (float*)d_out;           // [S,V]

    // ---- workspace layout: fp32 region then fp16 region (~158 MB) ----
    float* wsf  = (float*)d_ws;
    float* h    = wsf;  wsf += (size_t)S * D;      // hidden state (fp32 master)
    float* sc   = wsf;  wsf += (size_t)S * S;      // one head's scores
    float* tmpD = wsf;  wsf += (size_t)S * D;      // sublayer output

    _Float16* wsh   = (_Float16*)wsf;
    _Float16* h16   = wsh;  wsh += (size_t)S * D;
    _Float16* qkv16 = wsh;  wsh += (size_t)H * S * QN;  // per-head q|k|v (fp16)
    _Float16* sc16  = wsh;  wsh += (size_t)S * S;       // probs (fp16)
    _Float16* at16  = wsh;  wsh += (size_t)S * D;       // concat attention out
    _Float16* t4d16 = wsh;  wsh += (size_t)S * D4;      // MLP hidden (fp16)
    _Float16* qwT   = wsh;  wsh += (size_t)L * HS * QN; // [L][192][64]
    _Float16* owT   = wsh;  wsh += (size_t)L * D * D;   // [L][768][768]
    _Float16* w1T   = wsh;  wsh += (size_t)L * D * D4;  // [L][3072][768]
    _Float16* w2T   = wsh;  wsh += (size_t)L * D4 * D;  // [L][768][3072]
    _Float16* uwT   = wsh;  /* += (size_t)D * V; */     // [32000][768]

    dim3 blk(128);

    // ---- one-time fp32 -> fp16 transposed weight conversion ----
    convert_transpose_kernel<<<dim3(QN / 32, HS / 32, L), 256, 0, stream>>>(
        qkv_w, qwT, HS, QN, (long)HS * QN, (long)QN * HS);
    convert_transpose_kernel<<<dim3(D / 32, D / 32, L), 256, 0, stream>>>(
        out_w, owT, D, D, (long)D * D, (long)D * D);
    convert_transpose_kernel<<<dim3(D4 / 32, D / 32, L), 256, 0, stream>>>(
        w1, w1T, D, D4, (long)D * D4, (long)D4 * D);
    convert_transpose_kernel<<<dim3(D / 32, D4 / 32, L), 256, 0, stream>>>(
        w2, w2T, D4, D, (long)D4 * D, (long)D * D4);
    convert_transpose_kernel<<<dim3(V / 32, D / 32, 1), 256, 0, stream>>>(
        uw, uwT, D, V, 0, 0);

    embed_posn_kernel<<<S, 256, 0, stream>>>(x, embw, posn, h, h16, D);

    for (int l = 0; l < L; ++l) {
        // QKV: per-head [S,64] @ [64,192] -> fp16, batched over heads (grid.z)
        wmma_gemm_kernel<EP_NONE, true, true><<<dim3(QN / 64, S / 128, H), blk, 0, stream>>>(
            h16, qwT + (long)l * QN * HS, nullptr, qkv16,
            S, QN, HS, D, HS, QN,
            /*aBatch=*/HS, /*bBatch=*/0, /*cBatch=*/(long)S * QN, 0.f);

        for (int t = 0; t < H; ++t) {
            const _Float16* qh = qkv16 + (long)t * S * QN;
            // scores = (q @ k^T) * scale, causal mask fused (fp32 out for softmax)
            wmma_gemm_kernel<EP_SCALE_MASK, true, false><<<dim3(S / 64, S / 128, 1), blk, 0, stream>>>(
                qh + 0, qh + HS, nullptr, sc,
                S, S, HS, QN, QN, S, 0, 0, 0, scale);
            softmax_rows_kernel<<<S, 256, 0, stream>>>(sc, sc16, S);
            // attn_out[:, t*64:(t+1)*64] = probs @ v   (v row-major: non-TRANS B)
            wmma_gemm_kernel<EP_NONE, false, true><<<dim3(1, S / 128, 1), blk, 0, stream>>>(
                sc16, qh + 2 * HS, nullptr, at16 + (long)t * HS,
                S, HS, S, S, QN, D, 0, 0, 0, 0.f);
        }

        // out projection (fp32 out), then h += LN(.)
        wmma_gemm_kernel<EP_NONE, true, false><<<dim3(D / 64, S / 128, 1), blk, 0, stream>>>(
            at16, owT + (long)l * D * D, nullptr, tmpD,
            S, D, D, D, D, D, 0, 0, 0, 0.f);
        ln_residual_kernel<<<S, 256, 0, stream>>>(h, h16, tmpD,
            ln1_g + (long)l * D, ln1_b + (long)l * D, D);

        // MLP: gelu(h @ w1 + b1) (fp16) @ w2 + b2 (fp32), then h += LN(.)
        wmma_gemm_kernel<EP_BIAS_GELU, true, true><<<dim3(D4 / 64, S / 128, 1), blk, 0, stream>>>(
            h16, w1T + (long)l * D4 * D, b1 + (long)l * D4, t4d16,
            S, D4, D, D, D, D4, 0, 0, 0, 0.f);
        wmma_gemm_kernel<EP_BIAS, true, false><<<dim3(D / 64, S / 128, 1), blk, 0, stream>>>(
            t4d16, w2T + (long)l * D * D4, b2 + (long)l * D, tmpD,
            S, D, D4, D4, D4, D, 0, 0, 0, 0.f);
        ln_residual_kernel<<<S, 256, 0, stream>>>(h, h16, tmpD,
            ln2_g + (long)l * D, ln2_b + (long)l * D, D);
    }

    // unembed + softmax over vocabulary
    wmma_gemm_kernel<EP_BIAS, true, false><<<dim3(V / 64, S / 128, 1), blk, 0, stream>>>(
        h16, uwT, ub, out, S, V, D, D, D, V, 0, 0, 0, 0.f);
    softmax_rows_kernel<<<S, 256, 0, stream>>>(out, nullptr, V);
}
